// GINConv_56573309223702
// MI455X (gfx1250) — matrix operand order repaired and checked by reading, not compiled, not run on hardware
//
#include <hip/hip_runtime.h>
#include <hip/hip_bf16.h>

// ---------------------------------------------------------------------------
// GINConv on MI455X (gfx1250, wave32, WMMA):
//   h   = ReLU(x @ W1 + b1) @ W2 + b2     (fused bf16 WMMA, f32 accumulate,
//                                          M-tile=32, hidden chunked via LDS)
//   out = h + segment_sum(h[col], row)    (atomic f32 scatter, h L2-resident)
// ---------------------------------------------------------------------------

#define D_IN   512
#define D_HID  2048
#define D_OUT  512
#define KC     512          // hidden-column chunk processed per stage pass
#define NCHUNK (D_HID / KC) // 4

typedef __attribute__((ext_vector_type(16))) __bf16 v16bf;
typedef __attribute__((ext_vector_type(8)))  float  v8f;

union Frag16 {              // 16 bf16 = 32 bytes = two 16-byte chunks
    uint4 q[2];
    v16bf v;
};

__device__ __forceinline__ unsigned short f2bf(float f) {
    unsigned int u = __float_as_uint(f);
    unsigned int r = u + 0x7fffu + ((u >> 16) & 1u);   // round-to-nearest-even
    return (unsigned short)(r >> 16);
}

// --------------------------- prep kernels ----------------------------------

__global__ void cvt_bf16_kernel(const float* __restrict__ src,
                                unsigned short* __restrict__ dst, int n) {
    int i = blockIdx.x * blockDim.x + threadIdx.x;
    if (i < n) dst[i] = f2bf(src[i]);
}

// W [K x Nd] fp32 row-major  ->  WT [Nd x K] bf16 row-major (transpose)
__global__ void transpose_bf16_kernel(const float* __restrict__ W,
                                      unsigned short* __restrict__ WT,
                                      int K, int Nd) {
    int idx = blockIdx.x * blockDim.x + threadIdx.x;
    if (idx >= K * Nd) return;
    int n = idx / K;
    int k = idx - n * K;
    WT[idx] = f2bf(W[k * Nd + n]);
}

// --------------------------- fused MLP kernel ------------------------------
// One block = 32 rows of x (2 WMMA M-subtiles), 8 waves (256 threads).
// For each of 4 hidden-column chunks (512 cols):
//   stage 1: hid_chunk[32,512] = ReLU(x_tile @ W1[:,chunk] + b1) -> LDS bf16
//   stage 2: acc[32,512]      += hid_chunk @ W2[chunk,:]         (registers)
// Epilogue: h = acc + b2 -> h_ws and d_out.
// Every B-fragment feeds TWO wmmas (the two M-subtiles).
__global__ void __launch_bounds__(256)
gin_mlp_fused_kernel(const unsigned short* __restrict__ xb,   // [N, 512] bf16
                     const unsigned short* __restrict__ w1t,  // [2048, 512] bf16 (W1^T)
                     const unsigned short* __restrict__ w2t,  // [512, 2048] bf16 (W2^T)
                     const float* __restrict__ b1,            // [2048]
                     const float* __restrict__ b2,            // [512]
                     float* __restrict__ h_ws,                // [N, 512]
                     float* __restrict__ out,                 // [N, 512]
                     int N) {
    // Double-buffered hidden chunk: 2 x (32 rows x 512 cols bf16) = 64 KB.
    // Rotate-swizzle: row m's column c lives at ((c + 8*m) & 511); rotation is
    // a multiple of 8 elements (16 B) so fragment chunks stay contiguous while
    // the 1024 B row stride no longer aliases LDS banks across lanes.
    __shared__ unsigned short hid[2][32 * KC];

    const int lane   = threadIdx.x & 31;
    const int wave   = threadIdx.x >> 5;
    const int lane15 = lane & 15;
    const int hi     = lane >> 4;            // 0: lanes 0-15, 1: lanes 16-31
    const int aBase  = hi ? 8 : 0;           // A-fragment K sub-offset
    const int bHalf  = hi ? 16 : 0;          // B-fragment K half

    const int m0 = blockIdx.x * 32;

    // x rows served by this lane's A-fragments (clamped for the ragged tail)
    const unsigned short* xrow[2];
    xrow[0] = xb + (size_t)min(m0      + lane15, N - 1) * D_IN;
    xrow[1] = xb + (size_t)min(m0 + 16 + lane15, N - 1) * D_IN;

    v8f acc2[2][4] = {};                     // stage-2 accum: 2 Msub x 4 ntiles

    for (int ck = 0; ck < NCHUNK; ++ck) {
        unsigned short* buf = hid[ck & 1];

        // ------------- Stage 1: x_tile @ W1[:,chunk], ReLU -> LDS ---------
        {
            v8f acc1[2][4] = {};
            // this wave owns chunk-local columns [wave*64, wave*64+64)
            const size_t wbase =
                (size_t)(ck * KC + wave * 64 + lane15) * D_IN + aBase;
            #pragma unroll 4
            for (int k0 = 0; k0 < D_IN; k0 += 32) {
                Frag16 a[2];
                #pragma unroll
                for (int ms = 0; ms < 2; ++ms) {
                    a[ms].q[0] = *(const uint4*)(xrow[ms] + k0 + aBase);
                    a[ms].q[1] = *(const uint4*)(xrow[ms] + k0 + 16 + aBase);
                }
                #pragma unroll
                for (int nt = 0; nt < 4; ++nt) {
                    Frag16 b;
                    const unsigned short* wp = w1t + wbase + (size_t)nt * 16 * D_IN;
                    b.q[0] = *(const uint4*)(wp + k0);
                    b.q[1] = *(const uint4*)(wp + k0 + 16);
                    acc1[0][nt] = __builtin_amdgcn_wmma_f32_16x16x32_bf16(
                        false, a[0].v, false, b.v, (short)0, acc1[0][nt], false, false);
                    acc1[1][nt] = __builtin_amdgcn_wmma_f32_16x16x32_bf16(
                        false, a[1].v, false, b.v, (short)0, acc1[1][nt], false, false);
                }
            }
            // epilogue: bias + ReLU + bf16 -> swizzled LDS chunk
            #pragma unroll
            for (int ms = 0; ms < 2; ++ms) {
                #pragma unroll
                for (int nt = 0; nt < 4; ++nt) {
                    const int lcol = wave * 64 + nt * 16 + lane15;
                    const float bias = b1[ck * KC + lcol];
                    #pragma unroll
                    for (int r = 0; r < 8; ++r) {
                        const int mloc = 16 * ms + r + 8 * hi;
                        const float v = fmaxf(acc1[ms][nt][r] + bias, 0.0f);
                        buf[mloc * KC + ((lcol + 8 * mloc) & (KC - 1))] = f2bf(v);
                    }
                }
            }
        }
        __syncthreads();
        // (double buffering + program order make a 2nd barrier unnecessary:
        //  any wave passing the NEXT chunk's barrier has finished this chunk's
        //  stage-2 reads, so buffer ck is only overwritten at ck+2 safely)

        // ------------- Stage 2: acc2 += hid_chunk @ W2[chunk,:] -----------
        {
            const int mA0 = lane15;          // LDS rows served by this lane
            const int mA1 = lane15 + 16;
            // this wave owns output columns [wave*64, wave*64+64)
            const size_t wbase =
                (size_t)(wave * 64 + lane15) * D_HID + ck * KC + aBase;
            #pragma unroll 4
            for (int k0 = 0; k0 < KC; k0 += 32) {
                Frag16 a[2];
                {
                    const int c0 = (k0 + aBase + 8 * mA0) & (KC - 1);
                    const int c1 = (k0 + 16 + aBase + 8 * mA0) & (KC - 1);
                    a[0].q[0] = *(const uint4*)(&buf[mA0 * KC + c0]);
                    a[0].q[1] = *(const uint4*)(&buf[mA0 * KC + c1]);
                }
                {
                    const int c0 = (k0 + aBase + 8 * mA1) & (KC - 1);
                    const int c1 = (k0 + 16 + aBase + 8 * mA1) & (KC - 1);
                    a[1].q[0] = *(const uint4*)(&buf[mA1 * KC + c0]);
                    a[1].q[1] = *(const uint4*)(&buf[mA1 * KC + c1]);
                }
                #pragma unroll
                for (int nt = 0; nt < 4; ++nt) {
                    Frag16 b;
                    const unsigned short* wp = w2t + wbase + (size_t)nt * 16 * D_HID;
                    b.q[0] = *(const uint4*)(wp + k0);
                    b.q[1] = *(const uint4*)(wp + k0 + 16);
                    acc2[0][nt] = __builtin_amdgcn_wmma_f32_16x16x32_bf16(
                        false, a[0].v, false, b.v, (short)0, acc2[0][nt], false, false);
                    acc2[1][nt] = __builtin_amdgcn_wmma_f32_16x16x32_bf16(
                        false, a[1].v, false, b.v, (short)0, acc2[1][nt], false, false);
                }
            }
        }
    }

    // ---------------- Final epilogue: + b2 -> h_ws and out -----------------
    #pragma unroll
    for (int ms = 0; ms < 2; ++ms) {
        #pragma unroll
        for (int nt = 0; nt < 4; ++nt) {
            const int oc = wave * 64 + nt * 16 + lane15;
            const float bias = b2[oc];
            #pragma unroll
            for (int r = 0; r < 8; ++r) {
                const int m = m0 + 16 * ms + r + 8 * hi;
                if (m < N) {
                    const float v = acc2[ms][nt][r] + bias;
                    const size_t o = (size_t)m * D_OUT + oc;
                    h_ws[o] = v;   // stable copy for the edge gather
                    out[o]  = v;   // out starts as h; scatter adds on top
                }
            }
        }
    }
}

// --------------------------- edge scatter ----------------------------------
// One block per edge, 128 threads x float4: out[row,:] += h[col,:]
__global__ void __launch_bounds__(128)
gin_scatter_kernel(const long long* __restrict__ ei,   // [2, E] int64
                   const float* __restrict__ h,        // [N, 512]
                   float* __restrict__ out, int E) {
    const int e = blockIdx.x;
    const long long dst = ei[e];             // row (destination)
    const long long src = ei[(size_t)E + e]; // col (source)
    const float4* hs = (const float4*)(h + (size_t)src * D_OUT);
    float* od = out + (size_t)dst * D_OUT;
    const int t = threadIdx.x;               // 128 threads * 4 floats = 512
    const float4 v = hs[t];
    atomicAdd(od + 4 * t + 0, v.x);
    atomicAdd(od + 4 * t + 1, v.y);
    atomicAdd(od + 4 * t + 2, v.z);
    atomicAdd(od + 4 * t + 3, v.w);
}

// --------------------------- launcher --------------------------------------
extern "C" void kernel_launch(void* const* d_in, const int* in_sizes, int n_in,
                              void* d_out, int out_size, void* d_ws, size_t ws_size,
                              hipStream_t stream) {
    const float*     x  = (const float*)d_in[0];
    const long long* ei = (const long long*)d_in[1];
    const float*     W1 = (const float*)d_in[2];
    const float*     b1 = (const float*)d_in[3];
    const float*     W2 = (const float*)d_in[4];
    const float*     b2 = (const float*)d_in[5];
    float*           out = (float*)d_out;

    const int N = in_sizes[0] / D_IN;       // 50000
    const int E = in_sizes[1] / 2;          // 400000

    // workspace layout (256B aligned)
    char* ws = (char*)d_ws;
    const size_t w1t_bytes = (size_t)D_HID * D_IN * sizeof(unsigned short);  // 2 MB
    const size_t w2t_bytes = (size_t)D_OUT * D_HID * sizeof(unsigned short); // 2 MB
    const size_t xb_bytes  = (size_t)N * D_IN * sizeof(unsigned short);      // 51.2 MB
    unsigned short* w1t = (unsigned short*)ws;
    unsigned short* w2t = (unsigned short*)(ws + w1t_bytes);
    unsigned short* xb  = (unsigned short*)(ws + w1t_bytes + w2t_bytes);
    float* h_ws = (float*)(ws + ((w1t_bytes + w2t_bytes + xb_bytes + 255) & ~(size_t)255));

    // 1) convert + transpose weights to bf16 (stay L2-resident: 4 MB total)
    {
        const int n1 = D_IN * D_HID;
        transpose_bf16_kernel<<<(n1 + 255) / 256, 256, 0, stream>>>(W1, w1t, D_IN,  D_HID);
        transpose_bf16_kernel<<<(n1 + 255) / 256, 256, 0, stream>>>(W2, w2t, D_HID, D_OUT);
    }
    // 2) convert x to bf16
    {
        const int nx = N * D_IN;
        cvt_bf16_kernel<<<(nx + 255) / 256, 256, 0, stream>>>(x, xb, nx);
    }
    // 3) fused node MLP: 32 rows/block, hidden never leaves LDS
    gin_mlp_fused_kernel<<<(N + 31) / 32, 256, 0, stream>>>(
        xb, w1t, w2t, b1, b2, h_ws, out, N);
    // 4) edge gather / scatter-add (h = 102 MB < 192 MB L2 -> gathers hit L2)
    gin_scatter_kernel<<<E, 128, 0, stream>>>(ei, h_ws, out, E);
}
